// PPNet_50672024159104
// MI455X (gfx1250) — matrix operand order repaired and checked by reading, not compiled, not run on hardware
//
#include <hip/hip_runtime.h>
#include <hip/hip_bf16.h>

typedef __attribute__((ext_vector_type(16))) _Float16 v16h;
typedef __attribute__((ext_vector_type(8)))  _Float16 v8h;
typedef __attribute__((ext_vector_type(8)))  float    v8f;

#define EPS_ 1e-4f
#define FT   72   // padded half-stride of f tile rows (144B: 16B-aligned, breaks bank conflicts)

static __device__ __forceinline__ v16h cat8(v8h lo, v8h hi) {
  return __builtin_shufflevector(lo, hi, 0,1,2,3,4,5,6,7,8,9,10,11,12,13,14,15);
}
// order-preserving float<->uint so atomicMax(uint) == float max (deterministic)
static __device__ __forceinline__ unsigned fenc(float f) {
  unsigned u = __float_as_uint(f);
  return (u & 0x80000000u) ? ~u : (u | 0x80000000u);
}
static __device__ __forceinline__ float fdec(unsigned k) {
  unsigned u = (k & 0x80000000u) ? (k ^ 0x80000000u) : ~k;
  return __uint_as_float(u);
}

// ---------------------------------------------------------------------------
// Prep: prototypes -> f16 [192][64] (B-frag row-major [n][k]),
//       conv weights -> f16 [64][160] (K padded 147->160 with zeros),
//       fsum[n] = ||p_n||^2, pooled keys = 0 (identity of fenc-max).
// ---------------------------------------------------------------------------
__global__ void prep_kernel(const float* __restrict__ feat_w,
                            const float* __restrict__ proto,
                            _Float16* __restrict__ pf16,
                            _Float16* __restrict__ wf16,
                            float*    __restrict__ fsum,
                            unsigned* __restrict__ pooled) {
  const int tid = threadIdx.x; // 256 threads, 1 block
  for (int i = tid; i < 192 * 64; i += 256) {
    int n = i >> 6, k = i & 63;
    float v = (n < 190) ? proto[n * 64 + k] : 0.0f;
    pf16[i] = (_Float16)v;
  }
  for (int i = tid; i < 64 * 160; i += 256) {
    int c = i / 160, k = i % 160;
    float v = (k < 147) ? feat_w[c * 147 + k] : 0.0f;
    wf16[i] = (_Float16)v;
  }
  for (int n = tid; n < 192; n += 256) {
    float s = 0.0f;
    if (n < 190)
      for (int k = 0; k < 64; ++k) { float v = proto[n * 64 + k]; s += v * v; }
    fsum[n] = s;
  }
  for (int i = tid; i < 16 * 192; i += 256) pooled[i] = 0u;
}

// ---------------------------------------------------------------------------
// Fused conv(im2row WMMA) + sigmoid + prototype WMMA + distance/log epilogue.
// Block = 128 threads (4 wave32); handles (b, 2 output rows, 16 ow) = 32 pixels
// = 2 WMMA M-tiles sharing one staged 11x67x3 input patch.
// ---------------------------------------------------------------------------
__global__ __launch_bounds__(128)
void fused_kernel(const float* __restrict__ x,
                  const float* __restrict__ feat_b,
                  const _Float16* __restrict__ pf16,
                  const _Float16* __restrict__ wf16,
                  const float* __restrict__ fsum,
                  unsigned* __restrict__ pooled,
                  float* __restrict__ acts) {
  __shared__ float xs[3 * 11 * 68];                   // staged input patch (zero-padded)
  __shared__ __align__(32) _Float16 a_lds[32 * 160];  // im2row, 32 pixels x K160
  __shared__ __align__(32) _Float16 ftile[32 * FT];   // sigmoid(conv), 32 x 64 (stride FT)
  __shared__ float ssq[32];                           // per-pixel ||f||^2

  const int tid = threadIdx.x;
  const int bx  = blockIdx.x;            // 8192 = 16 b * 64 oh-pairs * 8 ow-tiles
  const int b   = bx >> 9;
  const int rem = bx & 511;
  const int oh0 = (rem >> 3) << 1;       // first of 2 output rows
  const int owb = (rem & 7) << 4;        // 16 ow per tile
  const int ihb = oh0 * 4 - 3;           // 11 input rows cover both output rows
  const int iwb = owb * 4 - 3;

  // ---- stage 11x67x3 input region (with zero padding) ----
  const float* xb = x + (size_t)b * (3 * 512 * 512);
  for (int i = tid; i < 3 * 11 * 68; i += 128) {
    int ci = i / 748;               // 748 = 11*68
    int r  = (i % 748) / 68;
    int cx = i % 68;
    float v = 0.0f;
    int ih = ihb + r, iw = iwb + cx;
    if (cx < 67 && ih >= 0 && ih < 512 && iw >= 0 && iw < 512)
      v = xb[(ci * 512 + ih) * 512 + iw];
    xs[i] = v;
  }
  __syncthreads();

  // ---- im2row -> f16 A tiles (2 x 16 x 160), K padded with zeros ----
  for (int i = tid; i < 32 * 160; i += 128) {
    int pos = i / 160, k = i % 160;     // pos = mt*16 + pixel-in-row
    float v = 0.0f;
    if (k < 147) {
      int ci = k / 49, r = (k % 49) / 7, s = k % 7;
      int mt = pos >> 4, pr = pos & 15;
      v = xs[(ci * 11 + (mt << 2) + r) * 68 + pr * 4 + s];
    }
    a_lds[i] = (_Float16)v;
  }
  __syncthreads();

  const int lane = tid & 31;
  const int wv   = tid >> 5;        // wave 0..3 -> conv channel tile
  const int nsel = lane & 15;
  const int hi   = lane >> 4;       // half-wave select
  const int row  = lane & 15;       // A-frag row (M)

  // ---- conv as WMMA: two 16pix x 16ch tiles per wave, K = 5 x 32 ----
  const int ccol = (wv << 4) + nsel;    // output channel 0..63
  v8f acc[2] = {{}, {}};
  #pragma unroll
  for (int kb = 0; kb < 5; ++kb) {
    v16h bm = *(const v16h*)(wf16 + ccol * 160 + kb * 32 + hi * 16);
    #pragma unroll
    for (int mt = 0; mt < 2; ++mt) {
      int ab = ((mt << 4) + row) * 160 + kb * 32 + hi * 16;
      v16h a = cat8(*(const v8h*)&a_lds[ab], *(const v8h*)&a_lds[ab + 8]);
      acc[mt] = __builtin_amdgcn_wmma_f32_16x16x32_f16(false, a, false, bm,
                                                       (short)0, acc[mt], false, false);
    }
  }
  // bias + sigmoid (fast rcp) -> LDS f tiles
  {
    float bv = feat_b[ccol];
    #pragma unroll
    for (int mt = 0; mt < 2; ++mt) {
      #pragma unroll
      for (int r = 0; r < 8; ++r) {
        int m = (mt << 4) + (hi << 3) + r;
        float f = __builtin_amdgcn_rcpf(1.0f + __expf(-(acc[mt][r] + bv)));
        ftile[m * FT + ccol] = (_Float16)f;
      }
    }
  }
  __syncthreads();

  // ---- per-pixel sum of squares (from the same f16 values the GEMM uses) ----
  if (tid < 32) {
    float s = 0.0f;
    #pragma unroll 8
    for (int c = 0; c < 64; ++c) { float f = (float)ftile[tid * FT + c]; s += f * f; }
    ssq[tid] = s;
  }
  __syncthreads();

  // ---- prototype GEMM: 12 n-tiles x 2 M-tiles over 4 waves ----
  #pragma unroll
  for (int t = 0; t < 3; ++t) {
    int col = ((wv * 3 + t) << 4) + nsel;     // prototype index 0..191
    const _Float16* bp = pf16 + col * 64 + hi * 16;
    v16h b0 = *(const v16h*)bp;
    v16h b1 = *(const v16h*)(bp + 32);
    float fs = fsum[col];
    #pragma unroll
    for (int mt = 0; mt < 2; ++mt) {
      int ab0 = ((mt << 4) + row) * FT + hi * 16;
      v16h a0 = cat8(*(const v8h*)&ftile[ab0],      *(const v8h*)&ftile[ab0 + 8]);
      v16h a1 = cat8(*(const v8h*)&ftile[ab0 + 32], *(const v8h*)&ftile[ab0 + 40]);
      v8f cc = {};
      cc = __builtin_amdgcn_wmma_f32_16x16x32_f16(false, a0, false, b0, (short)0, cc, false, false);
      cc = __builtin_amdgcn_wmma_f32_16x16x32_f16(false, a1, false, b1, (short)0, cc, false, false);

      float av[8];
      float mx = -3.4e38f;
      #pragma unroll
      for (int r = 0; r < 8; ++r) {
        float d = ssq[(mt << 4) + (hi << 3) + r] - 2.0f * cc[r] + fs;
        float a = __logf((d + 1.0f) * __builtin_amdgcn_rcpf(d + EPS_));
        av[r] = a;
        mx = fmaxf(mx, a);
      }
      if (col < 190) {
        size_t ob = ((size_t)(b * 190 + col)) * 16384
                  + (size_t)(oh0 + mt) * 128 + owb + (hi << 3);
        *(float4*)(acts + ob)     = make_float4(av[0], av[1], av[2], av[3]);
        *(float4*)(acts + ob + 4) = make_float4(av[4], av[5], av[6], av[7]);
      }
      mx = fmaxf(mx, __shfl_xor(mx, 16, 32));   // combine M=0..7 and M=8..15 halves
      if (hi == 0 && col < 190)
        atomicMax(&pooled[b * 192 + col], fenc(mx));
    }
  }
}

// ---------------------------------------------------------------------------
// logits[b][cls] = sum_p pooled[b][p] * last_w[cls][p]
// ---------------------------------------------------------------------------
__global__ void logits_kernel(const float* __restrict__ last_w,
                              const unsigned* __restrict__ pooled,
                              float* __restrict__ out) {
  int t = threadIdx.x;
  if (t >= 16 * 19) return;
  int b = t / 19, cls = t % 19;
  float s = 0.0f;
  for (int p = 0; p < 190; ++p)
    s += fdec(pooled[b * 192 + p]) * last_w[cls * 190 + p];
  out[b * 19 + cls] = s;
}

extern "C" void kernel_launch(void* const* d_in, const int* in_sizes, int n_in,
                              void* d_out, int out_size, void* d_ws, size_t ws_size,
                              hipStream_t stream) {
  (void)in_sizes; (void)n_in; (void)out_size; (void)ws_size;
  const float* x      = (const float*)d_in[0];
  const float* feat_w = (const float*)d_in[1];
  const float* feat_b = (const float*)d_in[2];
  const float* proto  = (const float*)d_in[3];   // (190,64,1,1) flat
  const float* last_w = (const float*)d_in[4];
  float* out = (float*)d_out;                    // [logits 16*19][acts 16*190*128*128]

  char* wsb = (char*)d_ws;
  _Float16* pf16   = (_Float16*)(wsb);           //   0    .. 24576 : protos f16 [192][64]
  _Float16* wf16   = (_Float16*)(wsb + 24576);   // 24576 .. 45056 : conv w f16 [64][160]
  float*    fsum   = (float*)(wsb + 45056);      // 45056 .. 45824 : ||p||^2
  unsigned* pooled = (unsigned*)(wsb + 46080);   // 46080 .. 58368 : max keys [16][192]

  prep_kernel<<<1, 256, 0, stream>>>(feat_w, proto, pf16, wf16, fsum, pooled);
  fused_kernel<<<dim3(16 * 64 * 8), dim3(128), 0, stream>>>(
      x, feat_b, pf16, wf16, fsum, pooled, out + 16 * 19);
  logits_kernel<<<1, 320, 0, stream>>>(last_w, pooled, out);
}
